// BinaryResNetBlock_7370163880468
// MI455X (gfx1250) — compile-verified, matrix operand order checked by reading, not run on hardware
//
#include <hip/hip_runtime.h>
#include <stdint.h>

// ---------------------------------------------------------------------------
// Bi-Real binary residual block on gfx1250 (MI455X).
// Binary conv == int8 (+1/-1) GEMM -> V_WMMA_I32_16X16X64_IU8.
// Memory-bound: binarize activations to halo-padded NHWC int8 once (27 MB,
// L2 resident), branchless constant-offset im2col, BN stats fused into the
// conv epilogue.
// ---------------------------------------------------------------------------

#define NB    32
#define CH    32
#define HH    160
#define WW    160
#define HP    (HH + 2)           // padded height 162
#define WP    (WW + 2)           // padded width 162
#define HW    (HH * WW)          // 25600
#define CHW   (CH * HW)          // 819200
#define TOTAL (NB * CHW)         // 26214400
#define KTAPS   9
#define KSTEPS  5                       // ceil(9 taps / 2 taps-per-64K-step)
#define WPACK_DWORDS (2 * KSTEPS * 32 * 8)   // 2 Mtiles x 5 Ksteps x 32 lanes x 8 dwords
#define STAT_COUNT ((float)(NB * HW))   // reduction count per channel
#define BN_EPS 1e-5f

// halo border: per image, (HP*WP - HH*WW) = 644 pixels, 8 dwords each
#define HALO_DWORDS (NB * 644 * 8)      // 164,864

typedef __attribute__((ext_vector_type(8))) int v8i;

// workspace layout (bytes)
#define XBP_OFF     0ull            // int8 padded NHWC activations (26,873,856 B)
#define Y1_OFF      26873856ull     // conv1 raw output, f32 NCHW (104,857,600 B)
#define INNER_OFF   131731456ull    // inner residual, f32 NCHW   (104,857,600 B)
#define WPACK1_OFF  236589056ull    // packed sign(W1), A-layout  (10,240 B)
#define WPACK2_OFF  236599296ull    // packed sign(W2)            (10,240 B)
#define STATS_OFF   236609536ull    // small float scratch
// stats float indices
#define S_SCALE1 0
#define S_SCALE2 1
#define S_SUM1   2     // 2..33
#define S_SQ1    34    // 34..65
#define S_SUM2   66    // 66..97
#define S_SQ2    98    // 98..129
#define S_A1     130
#define S_B1     162
#define S_A2     194
#define S_B2     226

// ---------------------------------------------------------------------------
// A-matrix (16x64 8-bit) VGPR byte layout (ISA 7.12.2):
//   lane L<16 : M=L,    dword v holds K = (v/2)*16 + (v&1)*4 + b
//   lane L>=16: M=L-16, dword v holds K = (v/2)*16 + 8 + (v&1)*4 + b
// K-order: k = tap*32 + ci  (tap 0..8, padded to 10 over 5 K-steps).
// ---------------------------------------------------------------------------

__global__ void pack_weights(const float* __restrict__ w1,
                             const float* __restrict__ w2,
                             float* __restrict__ stats,
                             int* __restrict__ wp1, int* __restrict__ wp2)
{
    const float* w = blockIdx.x ? w2 : w1;
    int*        wp = blockIdx.x ? wp2 : wp1;
    int tid = threadIdx.x;

    // scale = mean(|w|) over 32*32*9 = 9216 weights
    __shared__ float red[256];
    float s = 0.f;
    for (int i = tid; i < CH * CH * 9; i += 256) s += fabsf(w[i]);
    red[tid] = s;
    __syncthreads();
    for (int off = 128; off > 0; off >>= 1) {
        if (tid < off) red[tid] += red[tid + off];
        __syncthreads();
    }
    if (tid == 0) stats[blockIdx.x ? S_SCALE2 : S_SCALE1] = red[0] / (float)(CH * CH * 9);

    // pack sign(w) into A-matrix byte layout: dword index = ((mt*5+ks)*32+lane)*8+v
    for (int i = tid; i < WPACK_DWORDS; i += 256) {
        int v    = i & 7;
        int lane = (i >> 3) & 31;
        int g    = i >> 8;              // mt*KSTEPS + ks
        int ks   = g % KSTEPS;
        int mt   = g / KSTEPS;
        int m    = mt * 16 + (lane & 15);
        int half = lane >> 4;
        int kst  = ((v >> 1) << 4) + (half << 3) + ((v & 1) << 2);
        unsigned dw = 0u;
        for (int b = 0; b < 4; ++b) {
            int kl  = kst + b;
            int tap = ks * 2 + (kl >> 5);
            int ci  = kl & 31;
            int sb  = 0;
            if (tap < KTAPS) {
                float wv = w[(m * CH + ci) * 9 + tap];   // OIHW flat: (co*32+ci)*9 + tap
                sb = (wv > 0.f) ? 1 : ((wv < 0.f) ? -1 : 0);
            }
            dw |= ((unsigned)(unsigned char)(signed char)sb) << (8 * b);
        }
        wp[i] = (int)dw;
    }
}

// zero the halo border of the padded NHWC int8 buffer (dword granular)
__global__ void zero_halo(int* __restrict__ xbp4)
{
    int idx = blockIdx.x * 256 + threadIdx.x;
    if (idx >= HALO_DWORDS) return;
    int n   = idx / (644 * 8);
    int r   = idx % (644 * 8);
    int pix = r >> 3;
    int d   = r & 7;
    int y, x;
    if      (pix < 162)  { y = 0;          x = pix; }
    else if (pix < 324)  { y = HP - 1;     x = pix - 162; }
    else if (pix < 484)  { y = 1 + pix - 324; x = 0; }
    else                 { y = 1 + pix - 484; x = WP - 1; }
    xbp4[((n * HP + y) * WP + x) * 8 + d] = 0;
}

// sign(x): NCHW f32 -> padded NHWC int8, 4 channels (one dword) per thread.
// Also zeroes the BN stat accumulators once per call.
__global__ void binarize_x(const float* __restrict__ x,
                           int* __restrict__ xbp4,
                           float* __restrict__ stats)
{
    if (blockIdx.x == 0 && threadIdx.x < 128)
        stats[S_SUM1 + threadIdx.x] = 0.f;   // zero sums/sumsq for both convs

    int idx = blockIdx.x * 256 + threadIdx.x;         // over TOTAL/4
    if (idx >= TOTAL / 4) return;
    int xw = idx % WW;
    int t  = idx / WW;
    int y  = t % HH;
    int t2 = t / HH;
    int n  = t2 % NB;
    int cg = t2 / NB;                                 // channel group 0..7
    unsigned dw = 0u;
    #pragma unroll
    for (int j = 0; j < 4; ++j) {
        float v = x[((n * CH + cg * 4 + j) * HH + y) * WW + xw];
        int sb = (v > 0.f) ? 1 : ((v < 0.f) ? -1 : 0);
        dw |= ((unsigned)(unsigned char)(signed char)sb) << (8 * j);
    }
    xbp4[((n * HP + y + 1) * WP + (xw + 1)) * 8 + cg] = (int)dw;
}

// ---------------------------------------------------------------------------
// Core conv: each wave -> 32 Cout x 16 pixels (one W-segment of one row).
// D = A(16x64 i8 weights) x B(64x16 i8 im2col), 5 K-steps x 2 M-tiles
// -> 10 v_wmma_i32_16x16x64_iu8 per wave. Branchless B: one base pointer,
// all offsets compile-time immediates, int2 (b64) loads.
// Weights (10,240 B) staged to LDS with async global->LDS DMA.
// ---------------------------------------------------------------------------
__global__ __launch_bounds__(128) void conv_bin_wmma(
    const signed char* __restrict__ xbp, const int* __restrict__ wpack,
    const float* __restrict__ stats, int scale_idx,
    float* __restrict__ y, float* __restrict__ gsum, float* __restrict__ gsq)
{
    __shared__ int   s_wp[WPACK_DWORDS];
    __shared__ float s_sum[CH], s_sq[CH];
    int tid = threadIdx.x;

    // async stage packed weights into LDS: 640 x 16B, round-robin over threads
    for (int i = tid; i < WPACK_DWORDS / 4; i += 128) {
        unsigned    ldsoff = (unsigned)(uintptr_t)(&s_wp[i * 4]);
        const void* g      = (const void*)(wpack + i * 4);
        asm volatile("global_load_async_to_lds_b128 %0, %1, off"
                     :: "v"(ldsoff), "v"(g) : "memory");
    }
    if (tid < CH) { s_sum[tid] = 0.f; s_sq[tid] = 0.f; }
    asm volatile("s_wait_asynccnt 0" ::: "memory");
    __syncthreads();

    int lane  = tid & 31;
    int wave  = tid >> 5;
    int bw    = blockIdx.x % (WW / 16);
    int bh    = (blockIdx.x / (WW / 16)) % (HH / 4);
    int n     = blockIdx.x / ((WW / 16) * (HH / 4));
    int w0    = bw * 16;
    int h     = bh * 4 + wave;          // each of 4 waves owns one output row
    int lanef = lane & 15;              // pixel within 16-wide tile (B/D column)
    int half  = lane >> 4;

    // one base pointer per lane; every B load is base + compile-time immediate.
    // (padded coords: output (h, w0+lanef) reads rows h..h+2, cols +kx, all in range)
    const signed char* base =
        xbp + (((n * HP + h) * WP + (w0 + lanef)) * CH + half * 8);

    __builtin_prefetch(base + 2 * WP * CH, 0, 0);   // hint bottom halo row

    v8i c0 = {0, 0, 0, 0, 0, 0, 0, 0};
    v8i c1 = c0;

    #pragma unroll
    for (int ks = 0; ks < KSTEPS; ++ks) {
        // A tiles (both Cout halves) from LDS
        v8i a0 = *(const v8i*)&s_wp[((0 * KSTEPS + ks) * 32 + lane) * 8];
        v8i a1 = *(const v8i*)&s_wp[((1 * KSTEPS + ks) * 32 + lane) * 8];

        // B (64x16): k = tap*32 + ci; this K-step covers taps 2ks, 2ks+1.
        // dword pair p -> bytes {ci0..ci0+7} of one tap, ci0 = half*8 + (p&1)*16.
        v8i b;
        #pragma unroll
        for (int p = 0; p < 4; ++p) {
            int tap = ks * 2 + (p >> 1);          // compile-time after unroll
            if (tap < KTAPS) {
                int ky = tap / 3, kx = tap % 3;
                int off = (ky * WP + kx) * CH + (p & 1) * 16;
                int2 d = *(const int2*)(base + off);
                b[2 * p]     = d.x;
                b[2 * p + 1] = d.y;
            } else {
                b[2 * p]     = 0;
                b[2 * p + 1] = 0;
            }
        }

        c0 = __builtin_amdgcn_wmma_i32_16x16x64_iu8(true, a0, true, b, c0, false, false);
        c1 = __builtin_amdgcn_wmma_i32_16x16x64_iu8(true, a1, true, b, c1, false, false);
    }

    // epilogue: scale, store, BN batch-stat accumulation
    float scale = stats[scale_idx];
    int   px    = w0 + lanef;
    float* yrow = y + ((n * CH) * HH + h) * WW + px;   // + co*HW per channel
    #pragma unroll
    for (int r = 0; r < 8; ++r) {
        int   co0 = r + half * 8;          // D layout: lane half selects M(+8)
        int   co1 = 16 + co0;
        float v0  = scale * (float)c0[r];
        float v1  = scale * (float)c1[r];
        yrow[co0 * HW] = v0;
        yrow[co1 * HW] = v1;
        atomicAdd(&s_sum[co0], v0); atomicAdd(&s_sq[co0], v0 * v0);
        atomicAdd(&s_sum[co1], v1); atomicAdd(&s_sq[co1], v1 * v1);
    }
    __syncthreads();
    if (tid < CH) { atomicAdd(&gsum[tid], s_sum[tid]); atomicAdd(&gsq[tid], s_sq[tid]); }
}

// mean/var -> affine form a*y + b (training-mode BN, biased variance)
__global__ void bn_finalize(const float* __restrict__ gsum, const float* __restrict__ gsq,
                            const float* __restrict__ gamma, const float* __restrict__ beta,
                            float* __restrict__ a_out, float* __restrict__ b_out)
{
    int c = threadIdx.x;
    if (c >= CH) return;
    float m    = gsum[c] / STAT_COUNT;
    float var  = gsq[c] / STAT_COUNT - m * m;
    float rstd = rsqrtf(var + BN_EPS);
    float a    = gamma[c] * rstd;
    a_out[c] = a;
    b_out[c] = beta[c] - m * a;
}

// out = a[c]*yconv + b[c] + resid for 4 channels of one pixel per thread;
// optionally re-binarize into the padded NHWC int8 buffer (one dword store).
__global__ void bn_apply_residual(const float* __restrict__ yconv,
                                  const float* __restrict__ resid,
                                  const float* __restrict__ a, const float* __restrict__ b,
                                  float* __restrict__ out,
                                  int* __restrict__ xbp4, int write_xb)
{
    int idx = blockIdx.x * 256 + threadIdx.x;         // over TOTAL/4
    if (idx >= TOTAL / 4) return;
    int xw = idx % WW;
    int t  = idx / WW;
    int y  = t % HH;
    int t2 = t / HH;
    int n  = t2 % NB;
    int cg = t2 / NB;
    unsigned dw = 0u;
    #pragma unroll
    for (int j = 0; j < 4; ++j) {
        int c  = cg * 4 + j;
        int gi = ((n * CH + c) * HH + y) * WW + xw;
        float v = yconv[gi] * a[c] + b[c] + resid[gi];
        out[gi] = v;
        int sb = (v > 0.f) ? 1 : ((v < 0.f) ? -1 : 0);
        dw |= ((unsigned)(unsigned char)(signed char)sb) << (8 * j);
    }
    if (write_xb)
        xbp4[((n * HP + y + 1) * WP + (xw + 1)) * 8 + cg] = (int)dw;
}

extern "C" void kernel_launch(void* const* d_in, const int* in_sizes, int n_in,
                              void* d_out, int out_size, void* d_ws, size_t ws_size,
                              hipStream_t stream)
{
    const float* x      = (const float*)d_in[0];
    const float* w1     = (const float*)d_in[1];
    const float* gamma1 = (const float*)d_in[2];
    const float* beta1  = (const float*)d_in[3];
    const float* w2     = (const float*)d_in[4];
    const float* gamma2 = (const float*)d_in[5];
    const float* beta2  = (const float*)d_in[6];

    char* ws = (char*)d_ws;
    signed char* xbp   = (signed char*)(ws + XBP_OFF);
    int*         xbp4  = (int*)(ws + XBP_OFF);
    float*       y1    = (float*)(ws + Y1_OFF);
    float*       inner = (float*)(ws + INNER_OFF);
    int*         wp1   = (int*)(ws + WPACK1_OFF);
    int*         wp2   = (int*)(ws + WPACK2_OFF);
    float*       stats = (float*)(ws + STATS_OFF);
    float*       out   = (float*)d_out;

    const int qblocks = (TOTAL / 4 + 255) / 256;             // 25,600
    const int hblocks = (HALO_DWORDS + 255) / 256;           // 644
    const int cblocks = NB * (HH / 4) * (WW / 16);           // 12,800

    // prep
    pack_weights<<<2, 256, 0, stream>>>(w1, w2, stats, wp1, wp2);
    zero_halo<<<hblocks, 256, 0, stream>>>(xbp4);
    binarize_x<<<qblocks, 256, 0, stream>>>(x, xbp4, stats);
    // conv1 + BN1 + residual
    conv_bin_wmma<<<cblocks, 128, 0, stream>>>(xbp, wp1, stats, S_SCALE1,
                                               y1, stats + S_SUM1, stats + S_SQ1);
    bn_finalize<<<1, 32, 0, stream>>>(stats + S_SUM1, stats + S_SQ1,
                                      gamma1, beta1, stats + S_A1, stats + S_B1);
    bn_apply_residual<<<qblocks, 256, 0, stream>>>(y1, x, stats + S_A1, stats + S_B1,
                                                   inner, xbp4, 1);
    // conv2 + BN2 + residual (conv result parked in d_out, rewritten in place)
    conv_bin_wmma<<<cblocks, 128, 0, stream>>>(xbp, wp2, stats, S_SCALE2,
                                               out, stats + S_SUM2, stats + S_SQ2);
    bn_finalize<<<1, 32, 0, stream>>>(stats + S_SUM2, stats + S_SQ2,
                                      gamma2, beta2, stats + S_A2, stats + S_B2);
    bn_apply_residual<<<qblocks, 256, 0, stream>>>(out, inner, stats + S_A2, stats + S_B2,
                                                   out, (int*)nullptr, 0);
}
